// VectorQuantizer_20830591386203
// MI455X (gfx1250) — matrix operand order repaired and checked by reading, not compiled, not run on hardware
//
#include <hip/hip_runtime.h>
#include <hip/hip_bf16.h>
#include <cfloat>

typedef __attribute__((ext_vector_type(16))) __bf16 v16bf;
typedef __attribute__((ext_vector_type(8)))  float  v8f;

#define KCODES 1024
#define DDIM   64
#define CHUNK  128            // codes per LDS chunk; double-buffered => 64 KB LDS
#define ROWS_PER_BLOCK 256    // 16 waves * 16 rows
#define THREADS 512

// ---------------- prep: split codebook fp32 -> bf16 hi/lo + ||e||^2 ----------------
__global__ void vq_prep(const float* __restrict__ E,
                        __bf16* __restrict__ gEhi, __bf16* __restrict__ gElo,
                        float* __restrict__ gee) {
  int row = blockIdx.x * blockDim.x + threadIdx.x;
  if (row >= KCODES) return;
  const float* src = E + (size_t)row * DDIM;
  __bf16* dh = gEhi + (size_t)row * DDIM;
  __bf16* dl = gElo + (size_t)row * DDIM;
  float ss = 0.f;
#pragma unroll
  for (int d = 0; d < DDIM; ++d) {
    float x = src[d];
    ss = fmaf(x, x, ss);
    __bf16 h = (__bf16)x;
    dh[d] = h;
    dl[d] = (__bf16)(x - (float)h);   // exact residual in fp32
  }
  gee[row] = ss;
}

// ---------------- main: WMMA distances + streaming argmin ----------------
__global__ __launch_bounds__(THREADS) void vq_main(
    const float* __restrict__ X, const float* __restrict__ E,
    const __bf16* __restrict__ gEhi, const __bf16* __restrict__ gElo,
    const float* __restrict__ gee,
    float* __restrict__ outQ, float* __restrict__ outIdx,
    float* __restrict__ partials) {
  // [buffer][hi/lo][code*DDIM] : 2*2*128*64*2B = 64 KB (WGP has 320 KB)
  __shared__ __align__(32) __bf16 sE[2][2][CHUNK * DDIM];
  __shared__ float sRed[THREADS / 32];

  const int tid  = threadIdx.x;
  const int lane = tid & 31;
  const int wave = tid >> 5;
  const int n16  = lane & 15;
  const int g    = lane >> 4;

  const int m0    = blockIdx.x * ROWS_PER_BLOCK + wave * 16;
  const int myRow = m0 + n16;

  // Async chunk prefetch: global -> LDS directly (no VGPR round trip),
  // tracked by ASYNCcnt.  16 KB per matrix per chunk = 2 b128 per thread.
  auto issue_chunk = [&](int c, int buf) {
    const uint64_t gh = (uint64_t)(uintptr_t)(gEhi + (size_t)c * CHUNK * DDIM);
    const uint64_t gl = (uint64_t)(uintptr_t)(gElo + (size_t)c * CHUNK * DDIM);
    const uint32_t lh = (uint32_t)(uintptr_t)(&sE[buf][0][0]);
    const uint32_t ll = (uint32_t)(uintptr_t)(&sE[buf][1][0]);
#pragma unroll
    for (int r = 0; r < 2; ++r) {
      const uint32_t off = (uint32_t)(tid * 16 + r * (THREADS * 16));
      asm volatile("global_load_async_to_lds_b128 %0, %1, off"
                   :: "v"(lh + off), "v"(gh + off) : "memory");
      asm volatile("global_load_async_to_lds_b128 %0, %1, off"
                   :: "v"(ll + off), "v"(gl + off) : "memory");
    }
  };

  // Load 16x64 fp32 A tile once, split to bf16 hi/lo in WMMA A-layout.
  // A 16x32 bf16 layout: lane holds row (lane&15); element e -> K=16g+e.
  // D index = 32*s + K.
  v16bf a_hi[2], a_lo[2];
  {
    const float* xrow = X + (size_t)myRow * DDIM;
#pragma unroll
    for (int s = 0; s < 2; ++s) {
      const int d0 = 32 * s + 16 * g;
      float4 c0 = *(const float4*)(xrow + d0);
      float4 c1 = *(const float4*)(xrow + d0 + 4);
      float4 c2 = *(const float4*)(xrow + d0 + 8);
      float4 c3 = *(const float4*)(xrow + d0 + 12);
      float f[16] = {c0.x, c0.y, c0.z, c0.w, c1.x, c1.y, c1.z, c1.w,
                     c2.x, c2.y, c2.z, c2.w, c3.x, c3.y, c3.z, c3.w};
#pragma unroll
      for (int e = 0; e < 16; ++e) {
        __bf16 h = (__bf16)f[e];
        a_hi[s][e] = h;
        a_lo[s][e] = (__bf16)(f[e] - (float)h);
      }
    }
  }

  float best[8];
  int   bidx[8];
#pragma unroll
  for (int r = 0; r < 8; ++r) { best[r] = FLT_MAX; bidx[r] = 0; }

  const int NCHUNK = KCODES / CHUNK;
  issue_chunk(0, 0);
  for (int c = 0; c < NCHUNK; ++c) {
    const int buf = c & 1;
    asm volatile("s_wait_asynccnt 0x0" ::: "memory");  // my async writes done
    __syncthreads();                                   // everyone's writes visible
    if (c + 1 < NCHUNK) issue_chunk(c + 1, buf ^ 1);   // prefetch next chunk

    const __bf16* bhBase = &sE[buf][0][0];
    const __bf16* blBase = &sE[buf][1][0];

    for (int j = 0; j < CHUNK / 16; ++j) {
      const int crow = j * 16 + n16;  // local code = this lane's B column
      // B 32x16 bf16 layout: lane holds column n16, 16 contiguous K at 16g.
      v16bf b_hi[2], b_lo[2];
#pragma unroll
      for (int s = 0; s < 2; ++s) {
        const int off = crow * DDIM + 32 * s + 16 * g;
        b_hi[s] = *(const v16bf*)(bhBase + off);
        b_lo[s] = *(const v16bf*)(blBase + off);
      }
      // Two independent accumulator chains (3 WMMAs each) for ILP:
      // x.e in split precision: hi*hi + lo*hi + hi*lo  (~fp32 accuracy)
      v8f acc0 = {}, acc1 = {};
      acc0 = __builtin_amdgcn_wmma_f32_16x16x32_bf16(false, a_hi[0], false, b_hi[0], (short)0, acc0, false, false);
      acc1 = __builtin_amdgcn_wmma_f32_16x16x32_bf16(false, a_hi[0], false, b_lo[0], (short)0, acc1, false, false);
      acc0 = __builtin_amdgcn_wmma_f32_16x16x32_bf16(false, a_lo[0], false, b_hi[0], (short)0, acc0, false, false);
      acc1 = __builtin_amdgcn_wmma_f32_16x16x32_bf16(false, a_hi[1], false, b_hi[1], (short)0, acc1, false, false);
      acc0 = __builtin_amdgcn_wmma_f32_16x16x32_bf16(false, a_hi[1], false, b_lo[1], (short)0, acc0, false, false);
      acc1 = __builtin_amdgcn_wmma_f32_16x16x32_bf16(false, a_lo[1], false, b_hi[1], (short)0, acc1, false, false);

      const int   code = c * CHUNK + j * 16 + n16;
      const float eev  = gee[code];   // 16 consecutive dwords/wave-group, L2-hot
#pragma unroll
      for (int r = 0; r < 8; ++r) {
        // argmin of ||x||^2 + ||e||^2 - 2 x.e == argmin of ||e||^2 - 2 x.e
        float cand = fmaf(-2.0f, acc0[r] + acc1[r], eev);
        if (cand < best[r]) { best[r] = cand; bidx[r] = code; }  // '<' keeps first min
      }
    }
  }

  // argmin across the 16 lanes of each half-group (C layout: VGPR r, lanes
  // of group g hold row m0 + 8g + r, columns n16=0..15). Tie -> lowest index.
#pragma unroll
  for (int r = 0; r < 8; ++r) {
#pragma unroll
    for (int mask = 1; mask < 16; mask <<= 1) {
      float ob = __shfl_xor(best[r], mask, 32);
      int   oi = __shfl_xor(bidx[r], mask, 32);
      if (ob < best[r] || (ob == best[r] && oi < bidx[r])) { best[r] = ob; bidx[r] = oi; }
    }
  }

  // Gather winning rows (fp32, coalesced float4/lane), write outputs, MSE partial.
  float lsum = 0.f;
#pragma unroll
  for (int r = 0; r < 8; ++r) {
    const int m   = m0 + 8 * g + r;
    const int idx = bidx[r];                    // identical across the 16 lanes
    if (n16 == 0) outIdx[m] = (float)idx;
    float4 q  = ((const float4*)(E + (size_t)idx * DDIM))[n16];
    float4 xi = ((const float4*)(X + (size_t)m * DDIM))[n16];
    ((float4*)(outQ + (size_t)m * DDIM))[n16] = q;  // quantized_st == quantized
    float dx = q.x - xi.x, dy = q.y - xi.y, dz = q.z - xi.z, dw = q.w - xi.w;
    lsum += dx * dx + dy * dy + dz * dz + dw * dw;
  }
#pragma unroll
  for (int off = 16; off > 0; off >>= 1) lsum += __shfl_xor(lsum, off, 32);
  if (lane == 0) sRed[wave] = lsum;
  __syncthreads();
  if (tid == 0) {
    float s = 0.f;
    for (int w = 0; w < THREADS / 32; ++w) s += sRed[w];
    partials[blockIdx.x] = s;
  }
}

// ---------------- finish: deterministic loss reduction ----------------
__global__ void vq_finish(const float* __restrict__ partials, int nparts,
                          float* __restrict__ outLoss, float invCount) {
  __shared__ float sRed[8];
  int tid = threadIdx.x;
  float s = 0.f;
  for (int i = tid; i < nparts; i += blockDim.x) s += partials[i];
  for (int off = 16; off > 0; off >>= 1) s += __shfl_xor(s, off, 32);
  if ((tid & 31) == 0) sRed[tid >> 5] = s;
  __syncthreads();
  if (tid == 0) {
    float t = 0.f;
    for (int w = 0; w < 8; ++w) t += sRed[w];
    float e = t * invCount;
    outLoss[0] = 0.25f * e;  // vq_loss = COMMITMENT_COST * e_latent
    outLoss[1] = e;          // e_latent_loss
    outLoss[2] = 0.f;        // codebook_loss
  }
}

extern "C" void kernel_launch(void* const* d_in, const int* in_sizes, int n_in,
                              void* d_out, int out_size, void* d_ws, size_t ws_size,
                              hipStream_t stream) {
  const float* X = (const float*)d_in[0];     // [B,T,D] fp32
  const float* E = (const float*)d_in[1];     // [K,D]   fp32
  const int N = in_sizes[0] / DDIM;           // 65536 rows

  float* outQ    = (float*)d_out;             // [N*D]
  float* outIdx  = outQ + (size_t)N * DDIM;   // [N] (indices as float)
  float* outLoss = outIdx + N;                // vq, e_latent, codebook

  char* ws = (char*)d_ws;
  __bf16* gEhi  = (__bf16*)(ws);              // 128 KB
  __bf16* gElo  = (__bf16*)(ws + 131072);     // 128 KB
  float*  gee   = (float*)(ws + 262144);      // 4 KB
  float*  parts = (float*)(ws + 266240);      // <=1 KB

  const int nblocks = N / ROWS_PER_BLOCK;     // 256

  vq_prep<<<(KCODES + 255) / 256, 256, 0, stream>>>(E, gEhi, gElo, gee);
  vq_main<<<nblocks, THREADS, 0, stream>>>(X, E, gEhi, gElo, gee, outQ, outIdx, parts);
  vq_finish<<<1, 256, 0, stream>>>(parts, nblocks, outLoss, 1.0f / ((float)N * (float)DDIM));
}
